// Attention_31404800868979
// MI455X (gfx1250) — matrix operand order repaired and checked by reading, not compiled, not run on hardware
//
#include <hip/hip_runtime.h>

// ---------------------------------------------------------------------------
// Problem constants (B,S,HID)=(2,2048,2048), (H,HKV,DH)=(16,16,128)
// ---------------------------------------------------------------------------
#define B_ 2
#define S_ 2048
#define HID_ 2048
#define H_ 16
#define HKV_ 16
#define DH_ 128
#define NQKV_ 6144          // (H + 2*HKV) * DH
#define MROWS_ 4096         // B*S

typedef __bf16 v16bf __attribute__((ext_vector_type(16)));
typedef __bf16 v8bf  __attribute__((ext_vector_type(8)));
typedef float  v8f   __attribute__((ext_vector_type(8)));
typedef int    v4i   __attribute__((ext_vector_type(4)));
typedef unsigned int v4u __attribute__((ext_vector_type(4)));
typedef int    v8i   __attribute__((ext_vector_type(8)));

static __device__ __forceinline__ v8f wmma_bf16(v16bf a, v16bf b, v8f c) {
    return __builtin_amdgcn_wmma_f32_16x16x32_bf16(
        /*neg_a=*/false, a, /*neg_b=*/false, b,
        /*c_mod=*/(short)0, c, /*reuse_a=*/false, /*reuse_b=*/false);
}

// ---------------------------------------------------------------------------
// Async global->LDS copy (16 bytes), ASYNCcnt-tracked.
// Builtin prototype (from clang diagnostic): (v4i AS1*, v4i AS3*, Ii, Ii).
// ---------------------------------------------------------------------------
typedef __attribute__((address_space(1))) v4i* gas_v4i_ptr;
typedef __attribute__((address_space(3))) v4i* lds_v4i_ptr;

static __device__ __forceinline__ void async_cp_b128(void* ldsp, const void* gp) {
#if __has_builtin(__builtin_amdgcn_global_load_async_to_lds_b128)
    __builtin_amdgcn_global_load_async_to_lds_b128(
        (gas_v4i_ptr)gp, (lds_v4i_ptr)ldsp, 0, 0);
#else
    unsigned lds32 = (unsigned)(uintptr_t)ldsp;       // LDS aperture: low 32 bits
    asm volatile("global_load_async_to_lds_b128 %0, %1, off"
                 :: "v"(lds32), "v"((unsigned long long)(uintptr_t)gp)
                 : "memory");
#endif
}

#if __has_builtin(__builtin_amdgcn_s_wait_asynccnt)
#define WAIT_ASYNC(n) __builtin_amdgcn_s_wait_asynccnt(n)
#else
#define WAIT_ASYNC(n) asm volatile("s_wait_asynccnt %0" :: "i"(n) : "memory")
#endif

#if __has_builtin(__builtin_amdgcn_s_wait_tensorcnt)
#define WAIT_TENSOR(n) __builtin_amdgcn_s_wait_tensorcnt(n)
#else
#define WAIT_TENSOR(n) asm volatile("s_wait_tensorcnt %0" :: "i"(n) : "memory")
#endif

// ---------------------------------------------------------------------------
// TDM: DMA a 2D bf16 tile (tileD1 rows x tileD0 elems, row stride strideElems)
// from global to LDS. D# layout per CDNA5 ISA.
// This toolchain's builtin is the 6-arg form:
//   (uint32x4 g0, int32x8 g1, int32x4 g2, int32x4 g3, int32x8 g4, i32 cpol)
// Issued by one wave; completion via TENSORcnt.
// ---------------------------------------------------------------------------
#if __has_builtin(__builtin_amdgcn_tensor_load_to_lds)
#define HAVE_TDM 1
static __device__ __forceinline__ void tdm_load_2d_bf16(
    void* ldsp, const void* gp, int tileD0, int tileD1, int strideElems)
{
    const unsigned lds_addr = (unsigned)(uintptr_t)ldsp;
    const unsigned long long ga = (unsigned long long)(uintptr_t)gp;

    v4u g0;
    g0.x = 1u;                                   // count=1 (valid), user mode
    g0.y = lds_addr;                             // D#.lds_addr (bytes)
    g0.z = (unsigned)ga;                         // global_addr[31:0]
    g0.w = (unsigned)(ga >> 32) | (2u << 30);    // global_addr[56:32] | type=2

    v8i g1;
    g1[0] = (1 << 16);                           // data_size=1 (2 bytes/elem)
    g1[1] = (tileD0 & 0xFFFF) << 16;             // tensor_dim0[15:0] @ bit48
    g1[2] = ((unsigned)tileD0 >> 16) | ((tileD1 & 0xFFFF) << 16); // dim0 hi | dim1 lo
    g1[3] = ((unsigned)tileD1 >> 16) | (tileD0 << 16);            // dim1 hi | tile_dim0
    g1[4] = tileD1;                              // tile_dim1 | tile_dim2=0
    g1[5] = strideElems;                         // tensor_dim0_stride[31:0]
    g1[6] = 0;                                   // stride0 hi | stride1 lo
    g1[7] = 0;                                   // stride1 hi
    v4i g2 = {0, 0, 0, 0};                       // 2D tensor: groups 2/3 unused
    v4i g3 = {0, 0, 0, 0};
    v8i g4 = {0, 0, 0, 0, 0, 0, 0, 0};
    __builtin_amdgcn_tensor_load_to_lds(g0, g1, g2, g3, g4, 0);
}
#else
#define HAVE_TDM 0
#endif

// ---------------------------------------------------------------------------
// fp32 -> bf16 convert (grid-stride)
// ---------------------------------------------------------------------------
__global__ void cvt_f32_bf16(const float* __restrict__ in, __bf16* __restrict__ out,
                             long long n) {
    long long i = (long long)blockIdx.x * blockDim.x + threadIdx.x;
    long long stride = (long long)gridDim.x * blockDim.x;
    for (; i < n; i += stride) out[i] = (__bf16)in[i];
}

// ---------------------------------------------------------------------------
// GEMM: C[M,N] = A[M,K] * W[N,K]^T   (A,W bf16 row-major; fp32 accumulate)
// Block = 256 threads = 8 waves (4 M-waves x 2 N-waves); block tile 64M x 128N.
// Operands double-buffered in LDS via ASYNCcnt-tracked global->LDS copies:
// stage(k+1) overlaps the 4 WMMAs of tile k.
// ROPE=true: fused rotary embedding on q/k heads, bf16 output (qkv buffer).
// ---------------------------------------------------------------------------
template <bool ROPE>
__global__ __launch_bounds__(256) void gemm_bf16_kernel(
    const __bf16* __restrict__ A, const __bf16* __restrict__ W,
    __bf16* __restrict__ Cbf, float* __restrict__ Cf,
    const float* __restrict__ cosp, const float* __restrict__ sinp,
    int M, int N, int K)
{
    __shared__ __bf16 ldsA[2][64 * 32];    // 4 KB per buffer
    __shared__ __bf16 ldsW[2][128 * 32];   // 8 KB per buffer

    const int tid   = threadIdx.x;
    const int wid   = tid >> 5;
    const int lane  = tid & 31;
    const int l15   = lane & 15;
    const int khalf = lane >> 4;

    const int mBlock = blockIdx.y * 64;
    const int nBlock = blockIdx.x * 128;
    const int mw = wid >> 1;               // 0..3 : wave's 16-row slice
    const int nw = wid & 1;                // 0..1 : wave's 64-col slice

    // cooperative stage of one 64x32 A tile + 128x32 W tile (3 b128/thread)
    const int arow = tid >> 2, acol = (tid & 3) * 8;
    const int wrow = tid >> 1, wcol = (tid & 1) * 16;
    const __bf16* gA = A + (size_t)(mBlock + arow) * K + acol;
    const __bf16* gW = W + (size_t)(nBlock + wrow) * K + wcol;

    v8f acc[4];
#pragma unroll
    for (int t = 0; t < 4; ++t)
#pragma unroll
        for (int i = 0; i < 8; ++i) acc[t][i] = 0.f;

    const int nk = K / 32;

    // prologue: stage tile 0
    async_cp_b128(&ldsA[0][arow * 32 + acol], gA);
    async_cp_b128(&ldsW[0][wrow * 32 + wcol], gW);
    async_cp_b128(&ldsW[0][wrow * 32 + wcol + 8], gW + 8);

    for (int i = 0; i < nk; ++i) {
        const int cur = i & 1;
        if (i + 1 < nk) {
            const int kb = (i + 1) * 32;
            async_cp_b128(&ldsA[cur ^ 1][arow * 32 + acol], gA + kb);
            async_cp_b128(&ldsW[cur ^ 1][wrow * 32 + wcol], gW + kb);
            async_cp_b128(&ldsW[cur ^ 1][wrow * 32 + wcol + 8], gW + kb + 8);
            WAIT_ASYNC(3);                 // tile i done; tile i+1 in flight
        } else {
            WAIT_ASYNC(0);
        }
        __syncthreads();

        // A fragment from LDS (16x32; this code's k order: 16 contiguous K per lane half)
        v16bf af;
        {
            const __bf16* p = &ldsA[cur][(mw * 16 + l15) * 32 + khalf * 16];
            v8bf lo = *(const v8bf*)p;
            v8bf hi = *(const v8bf*)(p + 8);
#pragma unroll
            for (int j = 0; j < 8; ++j) { af[j] = lo[j]; af[8 + j] = hi[j]; }
        }
#pragma unroll
        for (int t = 0; t < 4; ++t) {
            const __bf16* bp = &ldsW[cur][(nw * 64 + t * 16 + l15) * 32 + khalf * 16];
            v16bf bf = *(const v16bf*)bp;
            acc[t] = wmma_bf16(af, bf, acc[t]);
        }
        __syncthreads();                   // readers done before buf reuse
    }

    // Epilogue. D layout: VGPR r, lanes 0-15 -> M=r, lanes 16-31 -> M=8+r; N=lane%16
#pragma unroll
    for (int t = 0; t < 4; ++t) {
        const int nB = nBlock + nw * 64 + t * 16;
#pragma unroll
        for (int r = 0; r < 8; ++r) {
            const int m = mBlock + mw * 16 + r + 8 * khalf;
            const int n = nB + l15;
            float v = acc[t][r];
            if (ROPE) {
                // rotate-half partner lives in neighbor lane (col n^1)
                float pair = __shfl_xor(v, 1, 32);
                const int head = n >> 7;          // n / DH
                const int d    = n & (DH_ - 1);
                const int srow = m & (S_ - 1);    // m = b*S + s
                const float c  = cosp[srow * DH_ + d];
                const float sn = sinp[srow * DH_ + d];
                const float rh = (d & 1) ? pair : -pair;
                const float outv = (head < (H_ + HKV_)) ? (v * c + rh * sn) : v;
                Cbf[(size_t)m * N + n] = (__bf16)outv;
            } else {
                Cf[(size_t)m * N + n] = v;
            }
        }
    }
}

// ---------------------------------------------------------------------------
// Flash attention: grid = (S/128, H, B); block = 256 threads (8 waves).
// Wave w owns 16 query rows; K tile staged by the Tensor Data Mover (one
// descriptor per block, TENSORcnt-tracked); V staged transposed by VALU.
// Online softmax with shfl reductions; P transposed via per-wave LDS scratch.
// ---------------------------------------------------------------------------
__global__ __launch_bounds__(256) void attn_kernel(
    const __bf16* __restrict__ qkv, __bf16* __restrict__ obf)
{
    __shared__ __bf16 ldsK[32 * DH_];    // K block, row-major [kcol][dh]
    __shared__ __bf16 ldsVT[DH_ * 32];   // V block transposed [dh][kcol]
    __shared__ __bf16 ldsP[8][16 * 32];  // per-wave P scratch

    const int tid   = threadIdx.x;
    const int wid   = tid >> 5;
    const int lane  = tid & 31;
    const int l15   = lane & 15;
    const int khalf = lane >> 4;

    const int qTile = blockIdx.x * 128;
    const int h     = blockIdx.y;
    const int b     = blockIdx.z;

    const size_t ld = NQKV_;
    const __bf16* qbase = qkv + (size_t)(b * S_) * ld + (size_t)h * DH_;
    const __bf16* kbase = qkv + (size_t)(b * S_) * ld + (size_t)(H_ + h) * DH_;
    const __bf16* vbase = qkv + (size_t)(b * S_) * ld + (size_t)(H_ + HKV_ + h) * DH_;

    // ---- Load Q fragments (16 rows x 128 dh = 4 K-steps of A-layout) ----
    v16bf qf[4];
    {
        const __bf16* qp = qbase + (size_t)(qTile + wid * 16 + l15) * ld;
#pragma unroll
        for (int kb = 0; kb < 4; ++kb) {
            const __bf16* p = qp + kb * 32 + khalf * 16;
            v8bf lo = *(const v8bf*)p;
            v8bf hi = *(const v8bf*)(p + 8);
#pragma unroll
            for (int i = 0; i < 8; ++i) { qf[kb][i] = lo[i]; qf[kb][8 + i] = hi[i]; }
        }
    }

    float rowmax[8], rowsum[8];
    v8f oacc[8];
#pragma unroll
    for (int r = 0; r < 8; ++r) { rowmax[r] = -1e30f; rowsum[r] = 0.f; }
#pragma unroll
    for (int t = 0; t < 8; ++t)
#pragma unroll
        for (int i = 0; i < 8; ++i) oacc[t][i] = 0.f;

    const float scale = 0.08838834764831845f; // 1/sqrt(128)
    const int kcol = tid >> 3;               // staging: 8 threads per key row
    const int dh0  = (tid & 7) * 16;         // 16 elements per thread

    for (int kblk = 0; kblk < S_; kblk += 32) {
        __syncthreads();
        // ---- Stage K (TDM if available, else async copies); V transposed ----
        {
#if HAVE_TDM
            if (wid == 0) {   // one DMA descriptor per block; EXEC-independent
                tdm_load_2d_bf16(&ldsK[0], kbase + (size_t)kblk * ld,
                                 /*tileD0=*/DH_, /*tileD1=*/32, /*stride=*/NQKV_);
            }
#else
            const __bf16* ks = kbase + (size_t)(kblk + kcol) * ld + dh0;
            async_cp_b128(&ldsK[kcol * DH_ + dh0], ks);
            async_cp_b128(&ldsK[kcol * DH_ + dh0 + 8], ks + 8);
#endif
            const __bf16* vs = vbase + (size_t)(kblk + kcol) * ld + dh0;
            v8bf v0 = *(const v8bf*)vs;
            v8bf v1 = *(const v8bf*)(vs + 8);
#pragma unroll
            for (int i = 0; i < 8; ++i) {
                ldsVT[(dh0 + i) * 32 + kcol]     = v0[i];
                ldsVT[(dh0 + 8 + i) * 32 + kcol] = v1[i];
            }
            if (kblk + 32 < S_) {   // prefetch next block into cache hierarchy
                __builtin_prefetch(kbase + (size_t)(kblk + 32 + kcol) * ld + dh0, 0, 0);
                __builtin_prefetch(vbase + (size_t)(kblk + 32 + kcol) * ld + dh0, 0, 0);
            }
#if HAVE_TDM
            if (wid == 0) WAIT_TENSOR(0);
#else
            WAIT_ASYNC(0);
#endif
        }
        __syncthreads();

        // ---- S = Q * K^T  (16 q-rows x 32 k-cols, contract over dh=128) ----
        v8f sacc[2];
#pragma unroll
        for (int t = 0; t < 2; ++t)
#pragma unroll
            for (int i = 0; i < 8; ++i) sacc[t][i] = 0.f;
#pragma unroll
        for (int nsub = 0; nsub < 2; ++nsub) {
#pragma unroll
            for (int kb = 0; kb < 4; ++kb) {
                const __bf16* bp = &ldsK[(nsub * 16 + l15) * DH_ + kb * 32 + khalf * 16];
                v16bf kf = *(const v16bf*)bp;
                sacc[nsub] = wmma_bf16(qf[kb], kf, sacc[nsub]);
            }
        }

        // ---- Online softmax update ----
        float alpha[8];
#pragma unroll
        for (int r = 0; r < 8; ++r) {
            float s0 = sacc[0][r] * scale;
            float s1 = sacc[1][r] * scale;
            float mx = fmaxf(s0, s1);
            mx = fmaxf(mx, __shfl_xor(mx, 1, 32));
            mx = fmaxf(mx, __shfl_xor(mx, 2, 32));
            mx = fmaxf(mx, __shfl_xor(mx, 4, 32));
            mx = fmaxf(mx, __shfl_xor(mx, 8, 32));
            float mnew = fmaxf(rowmax[r], mx);
            float p0 = __expf(s0 - mnew);
            float p1 = __expf(s1 - mnew);
            alpha[r] = __expf(rowmax[r] - mnew);
            rowmax[r] = mnew;
            float ps = p0 + p1;
            ps += __shfl_xor(ps, 1, 32);
            ps += __shfl_xor(ps, 2, 32);
            ps += __shfl_xor(ps, 4, 32);
            ps += __shfl_xor(ps, 8, 32);
            rowsum[r] = rowsum[r] * alpha[r] + ps;
            // stash P (D-layout) into per-wave LDS for A-layout reload
            const int prow = r + 8 * khalf;
            ldsP[wid][prow * 32 + l15]      = (__bf16)p0;
            ldsP[wid][prow * 32 + 16 + l15] = (__bf16)p1;
        }
        asm volatile("s_wait_dscnt 0" ::: "memory"); // wave-local LDS RAW

        // ---- Reload P in WMMA A-layout (16x32) ----
        v16bf pf;
        {
            const __bf16* pp = &ldsP[wid][l15 * 32 + khalf * 16];
            v8bf lo = *(const v8bf*)pp;
            v8bf hi = *(const v8bf*)(pp + 8);
#pragma unroll
            for (int i = 0; i < 8; ++i) { pf[i] = lo[i]; pf[8 + i] = hi[i]; }
        }

        // ---- O = alpha*O + P * V  (contract over 32 keys) ----
#pragma unroll
        for (int dsub = 0; dsub < 8; ++dsub) {
            v8f c = oacc[dsub];
#pragma unroll
            for (int r = 0; r < 8; ++r) c[r] *= alpha[r];
            const __bf16* vp = &ldsVT[(dsub * 16 + l15) * 32 + khalf * 16];
            v16bf vf = *(const v16bf*)vp;
            oacc[dsub] = wmma_bf16(pf, vf, c);
        }
    }

    // ---- Normalize and write O (bf16, [B*S, H*DH]) ----
#pragma unroll
    for (int dsub = 0; dsub < 8; ++dsub) {
#pragma unroll
        for (int r = 0; r < 8; ++r) {
            const int row = qTile + wid * 16 + r + 8 * khalf;
            const int col = h * DH_ + dsub * 16 + l15;
            const float val = oacc[dsub][r] / rowsum[r];
            obf[(size_t)(b * S_ + row) * (size_t)(H_ * DH_) + col] = (__bf16)val;
        }
    }
}

// ---------------------------------------------------------------------------
// Launch
// ---------------------------------------------------------------------------
extern "C" void kernel_launch(void* const* d_in, const int* in_sizes, int n_in,
                              void* d_out, int out_size, void* d_ws, size_t ws_size,
                              hipStream_t stream) {
    const float* hs   = (const float*)d_in[0];
    const float* cosp = (const float*)d_in[1];
    const float* sinp = (const float*)d_in[2];
    const float* wqkv = (const float*)d_in[3];
    const float* wo   = (const float*)d_in[4];
    float* out = (float*)d_out;

    // workspace layout (bf16 elements)
    const long long n_hs   = (long long)B_ * S_ * HID_;        // 8,388,608
    const long long n_wqkv = (long long)NQKV_ * HID_;          // 12,582,912
    const long long n_wo   = (long long)HID_ * (H_ * DH_);     // 4,194,304
    const long long n_qkv  = (long long)B_ * S_ * NQKV_;       // 25,165,824

    __bf16* hs_bf   = (__bf16*)d_ws;
    __bf16* wqkv_bf = hs_bf + n_hs;
    __bf16* wo_bf   = wqkv_bf + n_wqkv;
    __bf16* qkv_bf  = wo_bf + n_wo;
    __bf16* o_bf    = qkv_bf + n_qkv;

    cvt_f32_bf16<<<1024, 256, 0, stream>>>(hs, hs_bf, n_hs);
    cvt_f32_bf16<<<1024, 256, 0, stream>>>(wqkv, wqkv_bf, n_wqkv);
    cvt_f32_bf16<<<1024, 256, 0, stream>>>(wo, wo_bf, n_wo);

    // QKV GEMM + fused RoPE -> bf16 qkv buffer
    dim3 g1(NQKV_ / 128, MROWS_ / 64);
    gemm_bf16_kernel<true><<<g1, 256, 0, stream>>>(
        hs_bf, wqkv_bf, qkv_bf, nullptr, cosp, sinp, MROWS_, NQKV_, HID_);

    // Flash attention -> bf16 context buffer
    dim3 g2(S_ / 128, H_, B_);
    attn_kernel<<<g2, 256, 0, stream>>>(qkv_bf, o_bf);

    // Output projection -> fp32 result
    dim3 g3(HID_ / 128, MROWS_ / 64);
    gemm_bf16_kernel<false><<<g3, 256, 0, stream>>>(
        o_bf, wo_bf, nullptr, out, nullptr, nullptr, MROWS_, HID_, HID_);
}
